// APNLayer_8787503087819
// MI455X (gfx1250) — matrix operand order repaired and checked by reading, not compiled, not run on hardware
//
#include <hip/hip_runtime.h>
#include <hip/hip_bf16.h>

typedef __bf16 bf16_t;
typedef __attribute__((ext_vector_type(16))) __bf16 v16bf;
typedef __attribute__((ext_vector_type(8)))  float  v8f;

#define B_DIM 4
#define L_DIM 4096
#define D_DIM 1024
#define H_DIM 16
#define HD    64
#define CHUNK 64
#define NCHUNK (L_DIM / CHUNK)

__device__ __forceinline__ v8f v8zero() {
  v8f z;
#pragma unroll
  for (int i = 0; i < 8; ++i) z[i] = 0.0f;
  return z;
}

__device__ __forceinline__ v8f wmma_bf16(v16bf a, v16bf b, v8f c) {
  return __builtin_amdgcn_wmma_f32_16x16x32_bf16(false, a, false, b, (short)0, c,
                                                 false, false);
}

// Async global -> LDS copy, 16B per lane (CDNA5 GLOBAL_LOAD_ASYNC_TO_LDS_B128,
// tracked by ASYNCcnt; no VGPR data path).
__device__ __forceinline__ void async_ld16(const void* gaddr, void* laddr) {
  unsigned lo = (unsigned)(size_t)laddr;     // addr[31:0] = LDS offset
  asm volatile("global_load_async_to_lds_b128 %0, %1, off"
               :: "v"(lo), "v"(gaddr) : "memory");
}

__device__ __forceinline__ void wait_async0() {
  asm volatile("s_wait_asynccnt 0x0" ::: "memory");
}

// A fragment: rows m0..m0+15, K k0..k0+31, row-major lds (stride ld).
// Lanes 0-15 hold K 0-15; lanes 16-31 hold K 16-31.
__device__ __forceinline__ v16bf load_a(const bf16_t* p, int ld, int m0, int k0, int lane) {
  const bf16_t* row = p + (m0 + (lane & 15)) * ld + k0;
  const int kb = (lane < 16) ? 0 : 16;
  v16bf a;
#pragma unroll
  for (int q = 0; q < 8; ++q) {
    int k = (q << 1) + kb;
    a[2 * q]     = row[k];
    a[2 * q + 1] = row[k + 1];
  }
  return a;
}

// B fragment where operand is staged TRANSPOSED in LDS: src[n][k] row-major,
// B[k][n] = src[n][k]. Per lane: 16 contiguous bf16 -> 2x ds_load_b128.
__device__ __forceinline__ v16bf load_bT(const bf16_t* p, int ld, int k0, int n0, int lane) {
  const bf16_t* row = p + (n0 + (lane & 15)) * ld + k0 + ((lane < 16) ? 0 : 16);
  v16bf b;
#pragma unroll
  for (int e = 0; e < 16; ++e) b[e] = row[e];
  return b;
}

// ---------------- pack kernels: tanh + hi/lo bf16 split ----------------
__global__ void pack_x_kernel(const float* __restrict__ x, bf16_t* __restrict__ xh,
                              bf16_t* __restrict__ xl, int n) {
  int stride = gridDim.x * blockDim.x;
  for (int i = blockIdx.x * blockDim.x + threadIdx.x; i < n; i += stride) {
    float a = tanhf(x[i]);
    bf16_t hi = (bf16_t)a;
    xh[i] = hi;
    xl[i] = (bf16_t)(a - (float)hi);
  }
}

__global__ void pack_w_kernel(const float* __restrict__ w, bf16_t* __restrict__ wh,
                              bf16_t* __restrict__ wl, int n) {
  int stride = gridDim.x * blockDim.x;
  for (int i = blockIdx.x * blockDim.x + threadIdx.x; i < n; i += stride) {
    float a = w[i];
    bf16_t hi = (bf16_t)a;
    wh[i] = hi;
    wl[i] = (bf16_t)(a - (float)hi);
  }
}

// ---------------- static GEMM: y = x_act @ W^T (split-bf16, 3 WMMA terms) ---
// Double-buffered async global->LDS staging: copy of K-tile k+1 overlaps the
// WMMAs of tile k; one barrier per K step.
__device__ __forceinline__ void gemm_issue(const bf16_t* xh, const bf16_t* xl,
                                           const bf16_t* wh, const bf16_t* wl,
                                           bf16_t* stg, int m0, int n0, int k0,
                                           int tid) {
  for (int ch = tid; ch < 256; ch += 128) {     // 64x32 bf16 = 256 x 16B chunks
    int r = ch >> 2, c = (ch & 3) * 8;
    int lo = r * 32 + c;
    async_ld16(&xh[(size_t)(m0 + r) * D_DIM + k0 + c], stg + lo);
    async_ld16(&xl[(size_t)(m0 + r) * D_DIM + k0 + c], stg + 2048 + lo);
    async_ld16(&wh[(size_t)(n0 + r) * D_DIM + k0 + c], stg + 4096 + lo);
    async_ld16(&wl[(size_t)(n0 + r) * D_DIM + k0 + c], stg + 6144 + lo);
  }
}

__launch_bounds__(128)
__global__ void gemm_kernel(const bf16_t* __restrict__ xh, const bf16_t* __restrict__ xl,
                            const bf16_t* __restrict__ wh, const bf16_t* __restrict__ wl,
                            float* __restrict__ y) {
  __shared__ bf16_t stage[2][4 * 2048];   // per stage: Ah | Al | Bh(n,k) | Bl(n,k)
  const int tid = threadIdx.x, lane = tid & 31, wv = tid >> 5;
  const int m0 = blockIdx.x * 64, n0 = blockIdx.y * 64;
  v8f acc[4];
#pragma unroll
  for (int nt = 0; nt < 4; ++nt) acc[nt] = v8zero();

  gemm_issue(xh, xl, wh, wl, stage[0], m0, n0, 0, tid);
  for (int k0 = 0; k0 < D_DIM; k0 += 32) {
    const int s = (k0 >> 5) & 1;
    wait_async0();          // own stage-k loads landed in LDS
    __syncthreads();        // all waves' stage-k landed; prev compute finished
    if (k0 + 32 < D_DIM)    // overlap next copy with this tile's WMMAs
      gemm_issue(xh, xl, wh, wl, stage[s ^ 1], m0, n0, k0 + 32, tid);

    const bf16_t* Ah = &stage[s][0];
    const bf16_t* Al = &stage[s][2048];
    const bf16_t* Bh = &stage[s][4096];
    const bf16_t* Bl = &stage[s][6144];
    v16bf ah = load_a(Ah, 32, 16 * wv, 0, lane);
    v16bf al = load_a(Al, 32, 16 * wv, 0, lane);
#pragma unroll
    for (int nt = 0; nt < 4; ++nt) {
      v16bf bh = load_bT(Bh, 32, 0, 16 * nt, lane);
      v16bf bl = load_bT(Bl, 32, 0, 16 * nt, lane);
      acc[nt] = wmma_bf16(ah, bh, acc[nt]);   // hi*hi
      acc[nt] = wmma_bf16(ah, bl, acc[nt]);   // hi*lo
      acc[nt] = wmma_bf16(al, bh, acc[nt]);   // lo*hi
    }
  }
  const int rlo = (lane < 16) ? 0 : 8, c0 = lane & 15;
#pragma unroll
  for (int nt = 0; nt < 4; ++nt)
#pragma unroll
    for (int r = 0; r < 8; ++r)
      y[(size_t)(m0 + 16 * wv + r + rlo) * D_DIM + n0 + nt * 16 + c0] = acc[nt][r];
}

// ---------------- chunkwise gated delta rule (one head per workgroup) ------
__launch_bounds__(128)
__global__ void delta_kernel(const bf16_t* __restrict__ xh, float* __restrict__ y,
                             const float* __restrict__ eta_p,
                             const float* __restrict__ laml_p) {
  __shared__ float  S[HD * HD];          // running state [kfeat][v], fp32
  __shared__ bf16_t S0t[HD * HD];        // bf16 S snapshot, TRANSPOSED [v][kfeat]
  __shared__ bf16_t Kc[CHUNK * HD];      // chunk of q==k (bf16) [t][feat]
  __shared__ bf16_t KDT[HD * CHUNK];     // lam^{63-t} * K, TRANSPOSED [feat][t]
  __shared__ float  RHS[CHUNK * HD];     // V, then rhs, then U (fp32) [t][v]
  __shared__ float  AF[CHUNK * CHUNK];   // strict-lower A for the solve
  __shared__ bf16_t MH[CHUNK * CHUNK];   // decayed+masked (QK^T) bf16 [i][j]
  __shared__ bf16_t UHt[HD * CHUNK];     // U bf16, TRANSPOSED [v][t]
  __shared__ float  lampow[CHUNK + 1];
  __shared__ float  scal[2];

  const int tid = threadIdx.x, lane = tid & 31, wv = tid >> 5;
  const int head = blockIdx.x;           // 0..63 = B*H
  const int b = head >> 4, h = head & 15;
  const size_t rowbase = (size_t)b * L_DIM;
  const int cb = h * HD;
  const int rlo = (lane < 16) ? 0 : 8, c0 = lane & 15;

  if (tid == 0) {
    float lam = 1.0f / (1.0f + __expf(-laml_p[0]));
    scal[0] = lam;
    scal[1] = eta_p[0] * (1.0f - lam) / (float)HD;
    float pw = 1.0f;
    for (int t = 0; t <= CHUNK; ++t) { lampow[t] = pw; pw *= lam; }
  }
  for (int i = tid; i < HD * HD; i += 128) S[i] = 0.0f;
  __syncthreads();
  const float beta = scal[1];

#pragma unroll 1
  for (int nc = 0; nc < NCHUNK; ++nc) {
    const size_t grow0 = rowbase + (size_t)nc * CHUNK;
    // K chunk: 512 chunks of 8 bf16 (16B); V chunk: 1024 chunks of 4 floats
    for (int ch = tid; ch < 512; ch += 128) {
      int t = ch >> 3, f = (ch & 7) * 8;
      *(uint4*)&Kc[t * HD + f] = *(const uint4*)&xh[(grow0 + t) * D_DIM + cb + f];
    }
    for (int ch = tid; ch < 1024; ch += 128) {
      int t = ch >> 4, f = (ch & 15) * 4;
      *(float4*)&RHS[t * HD + f] = *(const float4*)&y[(grow0 + t) * D_DIM + cb + f];
    }
    if (nc + 1 < NCHUNK) {               // warm L2/L0 for the next serial chunk
      __builtin_prefetch(&xh[(grow0 + CHUNK + (tid >> 1)) * D_DIM + cb], 0, 1);
      __builtin_prefetch(&y[(grow0 + CHUNK + (tid >> 1)) * D_DIM + cb], 0, 1);
    }
    for (int idx = tid; idx < HD * HD; idx += 128) {
      int kf = idx & 63, v = idx >> 6;
      S0t[v * HD + kf] = (bf16_t)S[kf * HD + v];     // transposed snapshot
    }
    __syncthreads();
    // Decay-scaled transposed K for the state update (Phase D A-operand)
    for (int idx = tid; idx < HD * CHUNK; idx += 128) {
      int f = idx >> 6, t = idx & 63;
      KDT[idx] = (bf16_t)(lampow[63 - t] * (float)Kc[t * HD + f]);
    }

    // Phase A: G1 = K @ S0, KKT = K @ K^T  (wave wv owns rows 16wv..16wv+15)
    v8f g1[4], kkt[4];
#pragma unroll
    for (int nt = 0; nt < 4; ++nt) { g1[nt] = v8zero(); kkt[nt] = v8zero(); }
#pragma unroll
    for (int kk = 0; kk < HD; kk += 32) {
      v16bf a = load_a(Kc, HD, 16 * wv, kk, lane);
#pragma unroll
      for (int nt = 0; nt < 4; ++nt) {
        g1[nt]  = wmma_bf16(a, load_bT(S0t, HD, kk, 16 * nt, lane), g1[nt]);
        kkt[nt] = wmma_bf16(a, load_bT(Kc,  HD, kk, 16 * nt, lane), kkt[nt]);
      }
    }
#pragma unroll
    for (int nt = 0; nt < 4; ++nt)
#pragma unroll
      for (int r = 0; r < 8; ++r) {
        int i = 16 * wv + r + rlo;
        int j = nt * 16 + c0;
        float kv  = kkt[nt][r];
        float dec = (i >= j) ? lampow[i - j] : 0.0f;
        AF[i * CHUNK + j] = (i > j) ? beta * dec * kv : 0.0f;
        MH[i * CHUNK + j] = (bf16_t)(dec * kv);      // inclusive mask (q==k)
        float pi = lampow[i + 1];
        RHS[i * HD + j] = beta * (RHS[i * HD + j] - pi * g1[nt][r]);
        g1[nt][r] = pi * g1[nt][r];                  // carry term for output
      }
    __syncthreads();

    // Phase B: blocked forward substitution (I + AF) U = RHS, in place.
#pragma unroll 1
    for (int blk = 0; blk < 4; ++blk) {
      if (blk > 0) {
        int jn = 16 * blk;
        for (int o = tid; o < 16 * HD; o += 128) {
          int i = jn + (o >> 6), v = o & 63;
          float s = 0.0f;
          for (int j = 0; j < jn; ++j) s += AF[i * CHUNK + j] * RHS[j * HD + v];
          RHS[i * HD + v] -= s;
        }
      }
      __syncthreads();
      if (tid < HD) {                 // columns independent: no inner barrier
        int v = tid;
#pragma unroll 1
        for (int r = 1; r < 16; ++r) {
          int i = 16 * blk + r;
          float s = RHS[i * HD + v];
          for (int j = 16 * blk; j < i; ++j) s -= AF[i * CHUNK + j] * RHS[j * HD + v];
          RHS[i * HD + v] = s;
        }
      }
      __syncthreads();
    }
    for (int idx = tid; idx < HD * CHUNK; idx += 128) {
      int v = idx >> 6, t = idx & 63;
      UHt[idx] = (bf16_t)RHS[t * HD + v];            // transposed U
    }
    __syncthreads();

    // Phase C: O = carry + (decayed QK^T) @ U; accumulate into y (disjoint cols)
    v8f oacc[4];
#pragma unroll
    for (int nt = 0; nt < 4; ++nt) oacc[nt] = g1[nt];
#pragma unroll
    for (int kk = 0; kk < CHUNK; kk += 32) {
      v16bf a = load_a(MH, CHUNK, 16 * wv, kk, lane);
#pragma unroll
      for (int nt = 0; nt < 4; ++nt)
        oacc[nt] = wmma_bf16(a, load_bT(UHt, CHUNK, kk, 16 * nt, lane), oacc[nt]);
    }
#pragma unroll
    for (int nt = 0; nt < 4; ++nt)
#pragma unroll
      for (int r = 0; r < 8; ++r) {
        int i = 16 * wv + r + rlo;
        size_t gi = (grow0 + i) * D_DIM + cb + nt * 16 + c0;
        y[gi] += oacc[nt][r];
      }

    // Phase D: S = lam^C * S + K^T diag(lam^{C-1-t}) U
    v8f sacc[4];
#pragma unroll
    for (int nt = 0; nt < 4; ++nt) sacc[nt] = v8zero();
#pragma unroll
    for (int kk = 0; kk < CHUNK; kk += 32) {
      v16bf a = load_a(KDT, CHUNK, 16 * wv, kk, lane);   // rows = kfeat
#pragma unroll
      for (int nt = 0; nt < 4; ++nt)
        sacc[nt] = wmma_bf16(a, load_bT(UHt, CHUNK, kk, 16 * nt, lane), sacc[nt]);
    }
    const float lamC = lampow[CHUNK];
#pragma unroll
    for (int nt = 0; nt < 4; ++nt)
#pragma unroll
      for (int r = 0; r < 8; ++r) {
        int m = 16 * wv + r + rlo;
        int v = nt * 16 + c0;
        S[m * HD + v] = lamC * S[m * HD + v] + sacc[nt][r];
      }
    __syncthreads();
  }
}

// ---------------- LayerNorm (in place on d_out) ----------------
__launch_bounds__(256)
__global__ void ln_kernel(float* __restrict__ y, const float* __restrict__ gamma,
                          const float* __restrict__ beta) {
  __shared__ float red[256];
  const int tid = threadIdx.x;
  const size_t row = blockIdx.x;
  float v[4];
  float s = 0.0f;
#pragma unroll
  for (int q = 0; q < 4; ++q) { v[q] = y[row * D_DIM + tid + q * 256]; s += v[q]; }
  red[tid] = s; __syncthreads();
  for (int off = 128; off > 0; off >>= 1) {
    if (tid < off) red[tid] += red[tid + off];
    __syncthreads();
  }
  float mean = red[0] * (1.0f / D_DIM);
  __syncthreads();
  float s2 = 0.0f;
#pragma unroll
  for (int q = 0; q < 4; ++q) { float d = v[q] - mean; s2 += d * d; }
  red[tid] = s2; __syncthreads();
  for (int off = 128; off > 0; off >>= 1) {
    if (tid < off) red[tid] += red[tid + off];
    __syncthreads();
  }
  float inv = rsqrtf(red[0] * (1.0f / D_DIM) + 1e-5f);
#pragma unroll
  for (int q = 0; q < 4; ++q) {
    int c = tid + q * 256;
    y[row * D_DIM + c] = (v[q] - mean) * inv * gamma[c] + beta[c];
  }
}

extern "C" void kernel_launch(void* const* d_in, const int* in_sizes, int n_in,
                              void* d_out, int out_size, void* d_ws, size_t ws_size,
                              hipStream_t stream) {
  (void)in_sizes; (void)n_in; (void)out_size; (void)ws_size;
  const float* x     = (const float*)d_in[0];
  const float* W     = (const float*)d_in[1];
  const float* gamma = (const float*)d_in[2];
  const float* betab = (const float*)d_in[3];
  const float* eta   = (const float*)d_in[4];
  const float* laml  = (const float*)d_in[5];
  float* y = (float*)d_out;

  char* ws = (char*)d_ws;
  const size_t NX = (size_t)B_DIM * L_DIM * D_DIM;   // 16,777,216
  const size_t NW = (size_t)D_DIM * D_DIM;           // 1,048,576
  bf16_t* xh = (bf16_t*)ws;
  bf16_t* xl = (bf16_t*)(ws + NX * 2);
  bf16_t* wh = (bf16_t*)(ws + NX * 4);
  bf16_t* wl = (bf16_t*)(ws + NX * 4 + NW * 2);

  pack_x_kernel<<<8192, 256, 0, stream>>>(x, xh, xl, (int)NX);
  pack_w_kernel<<<1024, 256, 0, stream>>>(W, wh, wl, (int)NW);

  dim3 gg(B_DIM * L_DIM / 64, D_DIM / 64);           // 256 x 16 workgroups
  gemm_kernel<<<gg, 128, 0, stream>>>(xh, xl, wh, wl, y);

  delta_kernel<<<B_DIM * H_DIM, 128, 0, stream>>>(xh, y, eta, laml);

  ln_kernel<<<B_DIM * L_DIM, 256, 0, stream>>>(y, gamma, betab);
}
